// GruBlock_42116449305244
// MI455X (gfx1250) — matrix operand order repaired and checked by reading, not compiled, not run on hardware
//
#include <hip/hip_runtime.h>
#include <hip/hip_bf16.h>

typedef __attribute__((ext_vector_type(16))) _Float16 v16h;
typedef __attribute__((ext_vector_type(8)))  _Float16 v8h;
typedef __attribute__((ext_vector_type(8)))  float    v8f;

#define T_LEN 1024
#define B_SZ  32
#define H_SZ  256
#define G3    768
#define HLD   264   // padded LDS row stride (halves) to avoid bank conflicts

__device__ __forceinline__ float sigmoid_f(float x) {
    return 1.0f / (1.0f + __expf(-x));
}
__device__ __forceinline__ float tanh_f(float x) {
    float e = __expf(2.0f * x);
    return (e - 1.0f) / (e + 1.0f);
}

__device__ __forceinline__ v16h cat16(v8h lo, v8h hi) {
    v16h r;
#pragma unroll
    for (int i = 0; i < 8; i++) { r[i] = lo[i]; r[8 + i] = hi[i]; }
    return r;
}

// A operand (16xK tile, 16-bit): lane l -> row m=l&15, half=l>>4.
// v16h elems 0..7  = k = kbase + half*8 + 0..7
// v16h elems 8..15 = k = kbase + 16 + half*8 + 0..7
__device__ __forceinline__ v16h load_a_tile(const _Float16* A, int ld,
                                            int rowBase, int kbase, int lane) {
    int m = lane & 15, half = lane >> 4;
    const _Float16* p = A + (size_t)(rowBase + m) * ld + kbase + half * 8;
    return cat16(*(const v8h*)p, *(const v8h*)(p + 16));
}

// B operand (Kx16 tile from row-major W (N x K)): lane l -> col n=nBase+(l&15),
// khalf=l>>4; elems 0..7 = k = kbase + khalf*8 + 0..7, elems 8..15 = k = kbase + 16 + khalf*8 + 0..7.
__device__ __forceinline__ v16h load_b_tile(const _Float16* W, int ld,
                                            int nBase, int kbase, int lane) {
    int n = nBase + (lane & 15), kh = lane >> 4;
    const _Float16* p = W + (size_t)n * ld + kbase + kh * 8;
    return cat16(*(const v8h*)p, *(const v8h*)(p + 16));
}

__device__ __forceinline__ v8f wmma_f16(v16h a, v16h b, v8f c) {
    return __builtin_amdgcn_wmma_f32_16x16x32_f16(false, a, false, b,
                                                  (short)0, c, false, false);
}

// ---------------- kernel 0: f32 -> f16 weight conversion ----------------
__global__ void k_f32_to_f16(const float* __restrict__ s,
                             _Float16* __restrict__ d, int n) {
    int i = blockIdx.x * 256 + threadIdx.x;
    if (i < n) d[i] = (_Float16)s[i];
}

// ---------------- kernel 1: LayerNorm + transpose to (T,B,H) f16 --------
// one wave per row; 8 waves per block
__global__ void k_layernorm(const float* __restrict__ x,
                            const float* __restrict__ g,
                            const float* __restrict__ bta,
                            _Float16* __restrict__ xn16) {
    int wave = threadIdx.x >> 5, lane = threadIdx.x & 31;
    int rr = blockIdx.x * 8 + wave;        // rr = b*T + t  (x is (B,T,H))
    int b = rr >> 10, t = rr & (T_LEN - 1);
    const float* row = x + (size_t)rr * H_SZ;
    float4 v0 = *(const float4*)(row + lane * 8);
    float4 v1 = *(const float4*)(row + lane * 8 + 4);
    float s = v0.x + v0.y + v0.z + v0.w + v1.x + v1.y + v1.z + v1.w;
    float ss = v0.x * v0.x + v0.y * v0.y + v0.z * v0.z + v0.w * v0.w +
               v1.x * v1.x + v1.y * v1.y + v1.z * v1.z + v1.w * v1.w;
#pragma unroll
    for (int m = 16; m >= 1; m >>= 1) {
        s  += __shfl_xor(s, m, 32);
        ss += __shfl_xor(ss, m, 32);
    }
    float mean = s * (1.0f / H_SZ);
    float var  = ss * (1.0f / H_SZ) - mean * mean;
    float inv  = rsqrtf(var + 1e-5f);
    float4 g0 = *(const float4*)(g + lane * 8);
    float4 g1 = *(const float4*)(g + lane * 8 + 4);
    float4 b0 = *(const float4*)(bta + lane * 8);
    float4 b1 = *(const float4*)(bta + lane * 8 + 4);
    v8h o;
    o[0] = (_Float16)((v0.x - mean) * inv * g0.x + b0.x);
    o[1] = (_Float16)((v0.y - mean) * inv * g0.y + b0.y);
    o[2] = (_Float16)((v0.z - mean) * inv * g0.z + b0.z);
    o[3] = (_Float16)((v0.w - mean) * inv * g0.w + b0.w);
    o[4] = (_Float16)((v1.x - mean) * inv * g1.x + b1.x);
    o[5] = (_Float16)((v1.y - mean) * inv * g1.y + b1.y);
    o[6] = (_Float16)((v1.z - mean) * inv * g1.z + b1.z);
    o[7] = (_Float16)((v1.w - mean) * inv * g1.w + b1.w);
    _Float16* dst = xn16 + ((size_t)t * B_SZ + b) * H_SZ + lane * 8;
    *(v8h*)dst = o;
}

// ---------------- kernel 2: input gate GEMMs, TRANSPOSED output ---------
// Computes xpT = W_ih . xn^T + b_ih, stored (T, 3H, B) in f16 so the scan
// reads its per-column gate inputs as single contiguous b128 loads.
// WMMA roles: M = gate cols (A = W_ih rows), N = flat rows t*B+b (B = xn rows).
// grid (2048 row-tiles, 12), block 256 = 8 waves, wave -> one (M,N) tile
__global__ void k_gemm_in(const _Float16* __restrict__ xn16,
                          const _Float16* __restrict__ w16f,
                          const _Float16* __restrict__ w16b,
                          const float* __restrict__ bihf,
                          const float* __restrict__ bihb,
                          _Float16* __restrict__ xpTf,
                          _Float16* __restrict__ xpTb) {
    int lane = threadIdx.x & 31, wave = threadIdx.x >> 5;
    int mtile = blockIdx.y * 8 + wave;     // 0..95 (gate-col tiles, both dirs)
    int dir = mtile / 48, gt = mtile % 48;
    const _Float16* W = dir ? w16b : w16f;
    const float* bias = dir ? bihb : bihf;
    _Float16* xpT = dir ? xpTb : xpTf;
    int gateBase = gt * 16;
    int rowBase = blockIdx.x * 16;         // 16 flat rows (t fixed, b half)
    int lcol = lane & 15, half = lane >> 4;
    v8f c;
#pragma unroll
    for (int i = 0; i < 8; i++) c[i] = bias[gateBase + i + 8 * half];
#pragma unroll
    for (int kb = 0; kb < 8; kb++) {
        v16h a  = load_a_tile(W, H_SZ, gateBase, kb * 32, lane);
        v16h bm = load_b_tile(xn16, H_SZ, rowBase, kb * 32, lane);
        c = wmma_f16(a, bm, c);
    }
    int tq = rowBase >> 5;        // timestep
    int bh = rowBase & 31;        // batch half base (0 or 16)
#pragma unroll
    for (int i = 0; i < 8; i++) {
        int m = gateBase + i + 8 * half;               // gate column
        xpT[((size_t)tq * G3 + m) * B_SZ + bh + lcol] = (_Float16)c[i];
    }
}

// ---------------- kernel 3: recurrent scan (1 block per direction) ------
// 512 threads = 16 waves; wave w owns hidden cols [16w,16w+16) for all 3 gates
// and both M-halves of the 32-row batch. W_hh (f16) lives in VGPRs (3x8 v16h),
// h lives in double-buffered LDS (f16, padded rows); h_prev (f32) in registers.
// Per step: 6 contiguous v8h gate-input loads issued first (overlap the WMMA
// chain), one barrier per step via LDS double buffering.
__global__ void __launch_bounds__(512, 1)
k_gru_scan(const _Float16* __restrict__ whh16f,
           const _Float16* __restrict__ whh16b,
           const float* __restrict__ bhhf,
           const float* __restrict__ bhhb,
           const _Float16* __restrict__ xpTf,
           const _Float16* __restrict__ xpTb,
           _Float16* __restrict__ yh16) {
    __shared__ _Float16 h16s[2][B_SZ * HLD];  // ~33 KB
    int dir = blockIdx.x;
    const _Float16* W   = dir ? whh16b : whh16f;
    const float* bhh    = dir ? bhhb : bhhf;
    const _Float16* xpT = dir ? xpTb : xpTf;
    int tid = threadIdx.x, lane = tid & 31, cb = tid >> 5;  // cb = 0..15
    int lcol = lane & 15, half = lane >> 4;

    // persistent recurrent weights in registers: 3 gates x 8 k-blocks
    v16h Breg[3][8];
#pragma unroll
    for (int g = 0; g < 3; g++)
#pragma unroll
        for (int kb = 0; kb < 8; kb++)
            Breg[g][kb] = load_b_tile(W, H_SZ, g * H_SZ + cb * 16, kb * 32, lane);

    float bb[3];
#pragma unroll
    for (int g = 0; g < 3; g++) bb[g] = bhh[g * H_SZ + cb * 16 + lcol];

    for (int i = tid; i < 2 * B_SZ * HLD; i += 512)
        h16s[0][i] = (_Float16)0.0f;    // zeros both buffers (contiguous)
    v8f hprev[2];
#pragma unroll
    for (int mt = 0; mt < 2; mt++)
#pragma unroll
        for (int i = 0; i < 8; i++) hprev[mt][i] = 0.0f;
    __syncthreads();

    // per-lane base into xpT for this wave's hidden column (halves)
    size_t xcolbase = (size_t)(cb * 16 + lcol) * B_SZ + 8 * half;

    int cur = 0;
    for (int t = 0; t < T_LEN; t++) {
        int tp = dir ? (T_LEN - 1 - t) : t;

        // -------- hoisted gate-input loads: 6 contiguous v8h (b128) loads
        const _Float16* xbase =
            xpT + (size_t)tp * G3 * B_SZ + xcolbase;  // + g*256*B + mt*16
        v8h xg[3][2];
#pragma unroll
        for (int g = 0; g < 3; g++)
#pragma unroll
            for (int mt = 0; mt < 2; mt++)
                xg[g][mt] = *(const v8h*)(xbase +
                                          (size_t)g * H_SZ * B_SZ + mt * 16);

        // -------- recurrent matmul: gh = h @ W_hh^T + b_hh
        const _Float16* hcur = &h16s[cur][0];
        v8f c[3][2];
#pragma unroll
        for (int g = 0; g < 3; g++)
#pragma unroll
            for (int mt = 0; mt < 2; mt++)
#pragma unroll
                for (int i = 0; i < 8; i++) c[g][mt][i] = bb[g];

#pragma unroll
        for (int kb = 0; kb < 8; kb++) {
            v16h a0 = load_a_tile(hcur, HLD, 0,  kb * 32, lane);
            v16h a1 = load_a_tile(hcur, HLD, 16, kb * 32, lane);
#pragma unroll
            for (int g = 0; g < 3; g++) {
                c[g][0] = wmma_f16(a0, Breg[g][kb], c[g][0]);
                c[g][1] = wmma_f16(a1, Breg[g][kb], c[g][1]);
            }
        }

        // -------- gates + state update; write NEXT buffer
        _Float16* hnxt = &h16s[cur ^ 1][0];
#pragma unroll
        for (int mt = 0; mt < 2; mt++) {
#pragma unroll
            for (int i = 0; i < 8; i++) {
                int b = mt * 16 + i + 8 * half;
                float r = sigmoid_f((float)xg[0][mt][i] + c[0][mt][i]);
                float z = sigmoid_f((float)xg[1][mt][i] + c[1][mt][i]);
                float n = tanh_f   ((float)xg[2][mt][i] + r * c[2][mt][i]);
                float hn = (1.0f - z) * n + z * hprev[mt][i];
                hprev[mt][i] = hn;
                _Float16 hh = (_Float16)hn;
                hnxt[b * HLD + cb * 16 + lcol] = hh;
                yh16[((size_t)tp * B_SZ + b) * (2 * H_SZ) + dir * H_SZ +
                     cb * 16 + lcol] = hh;
            }
        }
        __syncthreads();   // t's reads of buf[cur] done; t's writes of
                           // buf[cur^1] visible before t+1 reads them
        cur ^= 1;
    }
}

// ---------------- kernel 4: output projection + residual ----------------
// out(b,t,:) = x(b,t,:) + y(t*B+b,:) @ W_map^T + b_map ; K = 512
__global__ void k_gemm_out(const _Float16* __restrict__ yh16,
                           const _Float16* __restrict__ wmap16,
                           const float* __restrict__ bmap,
                           const float* __restrict__ x,
                           float* __restrict__ out) {
    int lane = threadIdx.x & 31, wave = threadIdx.x >> 5;
    int nt = blockIdx.y * 8 + wave;   // 0..15
    int rowBase = blockIdx.x * 16;
    int lcol = lane & 15, half = lane >> 4;
    int col = nt * 16 + lcol;
    float bv = bmap[col];
    v8f c;
#pragma unroll
    for (int i = 0; i < 8; i++) c[i] = bv;
#pragma unroll
    for (int kb = 0; kb < 16; kb++) {
        v16h a  = load_a_tile(yh16, 2 * H_SZ, rowBase, kb * 32, lane);
        v16h bm = load_b_tile(wmap16, 2 * H_SZ, nt * 16, kb * 32, lane);
        c = wmma_f16(a, bm, c);
    }
#pragma unroll
    for (int i = 0; i < 8; i++) {
        int r = rowBase + i + 8 * half;
        int t = r >> 5, b = r & (B_SZ - 1);
        size_t idx = ((size_t)b * T_LEN + t) * H_SZ + col;
        out[idx] = c[i] + x[idx];
    }
}

extern "C" void kernel_launch(void* const* d_in, const int* in_sizes, int n_in,
                              void* d_out, int out_size, void* d_ws, size_t ws_size,
                              hipStream_t stream) {
    (void)in_sizes; (void)n_in; (void)out_size; (void)ws_size;
    const float* x    = (const float*)d_in[0];
    const float* ln_g = (const float*)d_in[1];
    const float* ln_b = (const float*)d_in[2];
    const float* Wihf = (const float*)d_in[3];
    const float* Whhf = (const float*)d_in[4];
    const float* bihf = (const float*)d_in[5];
    const float* bhhf = (const float*)d_in[6];
    const float* Wihb = (const float*)d_in[7];
    const float* Whhb = (const float*)d_in[8];
    const float* bihb = (const float*)d_in[9];
    const float* bhhb = (const float*)d_in[10];
    const float* Wmap = (const float*)d_in[11];
    const float* bmap = (const float*)d_in[12];
    float* out = (float*)d_out;

    char* ws = (char*)d_ws;
    size_t off = 0;
    auto carve = [&](size_t bytes) -> void* {
        void* p = ws + off;
        off += (bytes + 255) & ~(size_t)255;
        return p;
    };
    const size_t NROW = (size_t)T_LEN * B_SZ;           // 32768
    _Float16* xn16   = (_Float16*)carve(NROW * H_SZ * 2);      // 16 MB
    _Float16* xpTf   = (_Float16*)carve(NROW * G3 * 2);        // 48 MB
    _Float16* xpTb   = (_Float16*)carve(NROW * G3 * 2);        // 48 MB
    _Float16* yh16   = (_Float16*)carve(NROW * 2 * H_SZ * 2);  // 32 MB
    _Float16* wihf16 = (_Float16*)carve((size_t)G3 * H_SZ * 2);
    _Float16* wihb16 = (_Float16*)carve((size_t)G3 * H_SZ * 2);
    _Float16* whhf16 = (_Float16*)carve((size_t)G3 * H_SZ * 2);
    _Float16* whhb16 = (_Float16*)carve((size_t)G3 * H_SZ * 2);
    _Float16* wmap16 = (_Float16*)carve((size_t)H_SZ * 2 * H_SZ * 2);

    const int NW = G3 * H_SZ;         // 196608
    const int NM = H_SZ * 2 * H_SZ;   // 131072
    k_f32_to_f16<<<(NW + 255) / 256, 256, 0, stream>>>(Wihf, wihf16, NW);
    k_f32_to_f16<<<(NW + 255) / 256, 256, 0, stream>>>(Wihb, wihb16, NW);
    k_f32_to_f16<<<(NW + 255) / 256, 256, 0, stream>>>(Whhf, whhf16, NW);
    k_f32_to_f16<<<(NW + 255) / 256, 256, 0, stream>>>(Whhb, whhb16, NW);
    k_f32_to_f16<<<(NM + 255) / 256, 256, 0, stream>>>(Wmap, wmap16, NM);

    k_layernorm<<<(int)(NROW / 8), 256, 0, stream>>>(x, ln_g, ln_b, xn16);

    k_gemm_in<<<dim3(NROW / 16, 12), 256, 0, stream>>>(
        xn16, wihf16, wihb16, bihf, bihb, xpTf, xpTb);

    k_gru_scan<<<2, 512, 0, stream>>>(whhf16, whhb16, bhhf, bhhb,
                                      xpTf, xpTb, yh16);

    k_gemm_out<<<dim3(NROW / 16, 2), 256, 0, stream>>>(
        yh16, wmap16, bmap, x, out);
}
